// ClippedGRU_33054068310145
// MI455X (gfx1250) — matrix operand order repaired and checked by reading, not compiled, not run on hardware
//
#include <hip/hip_runtime.h>
#include <hip/hip_bf16.h>
#include <stdint.h>

// ClippedGRU for MI455X (gfx1250, wave32, WMMA).
// - fused input projection + recurrent scan, bf16 WMMA 16x16x32, f32 state
// - per-WG weight slices resident in LDS (async data mover when available)
// - software-pipelined A-fragment global loads (register double buffering)
// - device-wide atomic barrier between time steps

typedef __attribute__((ext_vector_type(16))) __bf16 v16bf;
typedef __attribute__((ext_vector_type(8)))  __bf16 v8bf;
typedef __attribute__((ext_vector_type(8)))  float  v8f;
typedef int v4i_gcc __attribute__((vector_size(16)));   // matches builtin proto

#define B_   128
#define T_   1024
#define I_   256
#define H_   512
#define G3_  1536
#define NWG  32      // one workgroup per 16-column tile of H (512/16)

// ---- workspace layout (bytes, all offsets 256B aligned) ----
#define OFF_X    0ull
#define OFF_WIH  (OFF_X   + (size_t)B_ * T_ * I_ * 2)   // x in bf16: 64 MB
#define OFF_WHH  (OFF_WIH + (size_t)G3_ * I_ * 2)       // w_ih bf16
#define OFF_HF   (OFF_WHH + (size_t)G3_ * H_ * 2)       // w_hh bf16
#define OFF_HB   (OFF_HF  + (size_t)2 * B_ * H_ * 4)    // h double-buffer f32
#define OFF_CNT  (OFF_HB  + (size_t)2 * B_ * H_ * 2)    // h double-buffer bf16
// OFF_CNT: 256B barrier counter region

// ---------------- helpers ----------------

__device__ __forceinline__ float fast_sigmoid(float x) {
    return 1.0f / (1.0f + __expf(-x));
}
__device__ __forceinline__ float fast_tanh(float x) {
    float e = __expf(-2.0f * fabsf(x));
    float t = (1.0f - e) / (1.0f + e);
    return copysignf(t, x);
}

// A fragment (16x32 bf16, ISA 7.12.2): lane<16 -> row=lane, K = k0+0..7 and
// k0+16..23 ; lane>=16 -> row=lane-16, K = k0+8..15 and k0+24..31.
__device__ __forceinline__ v16bf load_a_g(const __bf16* __restrict__ base,
                                          size_t rowStride, int mBase, int k0,
                                          int lane) {
    int m = lane & 15, kh = lane >> 4;
    const __bf16* p = base + (size_t)(mBase + m) * rowStride + (size_t)(k0 + kh * 8);
    v16bf a;
    ((uint4*)&a)[0] = *(const uint4*)(p);
    ((uint4*)&a)[1] = *(const uint4*)(p + 16);
    return a;
}

// B fragment (32x16 bf16): lane<16 -> col=lane, K=k0+0..7 and k0+16..23;
// lane>=16 -> col=lane-16, K=k0+8..15 and k0+24..31.
__device__ __forceinline__ v16bf load_b_s(const __bf16* w, int Ks, int nBase,
                                          int k0, int lane) {
    int n = lane & 15, kh = lane >> 4;
    const __bf16* p = w + (size_t)(nBase + n) * Ks + (size_t)(k0 + kh * 8);
    v16bf b;
    ((uint4*)&b)[0] = *(const uint4*)(p);
    ((uint4*)&b)[1] = *(const uint4*)(p + 16);
    return b;
}

#define WMMA_BF16(A, Bm, C) \
    __builtin_amdgcn_wmma_f32_16x16x32_bf16(false, (A), false, (Bm), (short)0, (C), false, false)

// 16B global -> LDS copy: async data mover path when the toolchain has it.
#if __has_builtin(__builtin_amdgcn_global_load_async_to_lds_b128)
#define HAS_ASYNC_LDS 1
#else
#define HAS_ASYNC_LDS 0
#endif

__device__ __forceinline__ void g2l_b128(void* lds, const void* g) {
#if HAS_ASYNC_LDS
    __builtin_amdgcn_global_load_async_to_lds_b128(
        (__attribute__((address_space(1))) v4i_gcc*)(v4i_gcc*)(uintptr_t)g,
        (__attribute__((address_space(3))) v4i_gcc*)(v4i_gcc*)(uintptr_t)lds,
        0, 0);
#else
    *(uint4*)lds = *(const uint4*)g;
#endif
}

__device__ __forceinline__ void wait_async_lds() {
#if HAS_ASYNC_LDS
#if __has_builtin(__builtin_amdgcn_s_wait_asynccnt)
    __builtin_amdgcn_s_wait_asynccnt(0);
#else
    asm volatile("s_wait_asynccnt 0x0" ::: "memory");
#endif
#endif
}

// ---------------- setup kernels ----------------

__global__ void cvt_f32_to_bf16_x8(const float* __restrict__ src,
                                   __bf16* __restrict__ dst, int n8) {
    int i = blockIdx.x * blockDim.x + threadIdx.x;
    if (i >= n8) return;
    const float4* s = (const float4*)src + (size_t)i * 2;
    float4 a = s[0], b = s[1];
    v8bf o;
    o[0] = (__bf16)a.x; o[1] = (__bf16)a.y; o[2] = (__bf16)a.z; o[3] = (__bf16)a.w;
    o[4] = (__bf16)b.x; o[5] = (__bf16)b.y; o[6] = (__bf16)b.z; o[7] = (__bf16)b.w;
    *(v8bf*)(dst + (size_t)i * 8) = o;
}

__global__ void init_h_kernel(const float* __restrict__ h0,
                              float* __restrict__ hf,
                              __bf16* __restrict__ hb) {
    int i = blockIdx.x * blockDim.x + threadIdx.x;
    if (i >= B_ * H_) return;
    float v = h0[i & (H_ - 1)];
    hf[i] = v;
    hb[i] = (__bf16)v;
}

// ---------------- persistent scan kernel ----------------
// grid = NWG blocks (block jt owns H columns [16*jt, 16*jt+16));
// block = 256 threads = 8 waves; wave w owns B rows [16w, 16w+16).

__global__ __launch_bounds__(256)
void gru_scan_kernel(const __bf16* __restrict__ xbf,    // [B][T][I] bf16
                     const __bf16* __restrict__ wih,    // [3H][I]  bf16
                     const __bf16* __restrict__ whh,    // [3H][H]  bf16
                     const float*  __restrict__ b_ih,   // [3H]
                     const float*  __restrict__ b_hh,   // [3H]
                     float*        __restrict__ hf,     // 2x [B][H] f32
                     __bf16*       __restrict__ hb,     // 2x [B][H] bf16
                     float*        __restrict__ out,    // [B][T][H] + [B][H]
                     unsigned int* __restrict__ cnt) {
    extern __shared__ char smem[];
    __bf16* whh_s = (__bf16*)smem;                       // [48][512] (r,z,n tiles)
    __bf16* wih_s = (__bf16*)(smem + 48 * 512 * 2);      // [48][256]
    float*  biasS = (float*)(smem + 48 * 512 * 2 + 48 * 256 * 2); // 64 floats

    const int tid   = threadIdx.x;
    const int jt    = blockIdx.x;         // H column tile
    const int lane  = tid & 31;
    const int wave  = tid >> 5;
    const int mBase = wave * 16;

    // --- stage weight slices into LDS (once; async data mover if available) ---
    for (int idx = tid; idx < 48 * 64; idx += 256) {     // w_hh: 48 rows x 64 uint4
        int rowL = idx >> 6, q = idx & 63;
        int g = rowL >> 4, nl = rowL & 15;
        size_t grow = (size_t)(g * H_ + jt * 16 + nl);
        g2l_b128(((uint4*)whh_s) + (size_t)rowL * 64 + q,
                 ((const uint4*)whh) + grow * 64 + q);
    }
    for (int idx = tid; idx < 48 * 32; idx += 256) {     // w_ih: 48 rows x 32 uint4
        int rowL = idx >> 5, q = idx & 31;
        int g = rowL >> 4, nl = rowL & 15;
        size_t grow = (size_t)(g * H_ + jt * 16 + nl);
        g2l_b128(((uint4*)wih_s) + (size_t)rowL * 32 + q,
                 ((const uint4*)wih) + grow * 32 + q);
    }
    if (tid < 16) {
        int c = jt * 16 + tid;
        biasS[tid]      = b_ih[c]           + b_hh[c];           // r
        biasS[16 + tid] = b_ih[H_ + c]      + b_hh[H_ + c];      // z
        biasS[32 + tid] = b_ih[2 * H_ + c];                      // gi_n
        biasS[48 + tid] = b_hh[2 * H_ + c];                      // gh_n
    }
    wait_async_lds();
    __syncthreads();

    const float bR  = biasS[lane & 15];
    const float bZ  = biasS[16 + (lane & 15)];
    const float bIN = biasS[32 + (lane & 15)];
    const float bHN = biasS[48 + (lane & 15)];

    const int col   = (jt << 4) + (lane & 15);           // H column
    const int rbase = mBase + ((lane >> 4) << 3);        // C/D row base

    for (int t = 0; t < T_; ++t) {
        const float*  hprevF = hf + (size_t)(t & 1) * (B_ * H_);
        const __bf16* hprevB = hb + (size_t)(t & 1) * (B_ * H_);
        float*        hnextF = hf + (size_t)((t + 1) & 1) * (B_ * H_);
        __bf16*       hnextB = hb + (size_t)((t + 1) & 1) * (B_ * H_);

        // issue h_old loads early so they complete under the GEMMs
        float ho[8];
#pragma unroll
        for (int r = 0; r < 8; ++r)
            ho[r] = hprevF[(size_t)(rbase + r) * H_ + col];

        v8f accR, accZ, accIN, accHN;
#pragma unroll
        for (int r = 0; r < 8; ++r) {
            accR[r] = bR; accZ[r] = bZ; accIN[r] = bIN; accHN[r] = bHN;
        }

        // prefetch next step's x slice (global_prefetch_b8)
        if (t + 1 < T_) {
            const __bf16* pf = xbf + (size_t)(mBase + (lane & 15)) * ((size_t)T_ * I_)
                                   + (size_t)(t + 1) * I_ + (lane >> 4) * 64;
            __builtin_prefetch((const void*)pf, 0, 1);
        }

        const __bf16* xrow = xbf + (size_t)t * I_;       // row stride T*I

        // prologue A fragments: first gi tile and first gh tile both in flight
        v16bf a  = load_a_g(xrow,   (size_t)T_ * I_, mBase, 0, lane);
        v16bf ah = load_a_g(hprevB, (size_t)H_,      mBase, 0, lane);

        // ---- gi = x_t @ w_ih^T  (K = 256), A double-buffered ----
#pragma unroll 2
        for (int k0 = 0; k0 < I_; k0 += 32) {
            v16bf an = load_a_g(xrow, (size_t)T_ * I_, mBase,
                                (k0 + 32) & (I_ - 1), lane);
            v16bf b0 = load_b_s(wih_s, I_, 0,  k0, lane);
            accR  = WMMA_BF16(a, b0, accR);
            v16bf b1 = load_b_s(wih_s, I_, 16, k0, lane);
            accZ  = WMMA_BF16(a, b1, accZ);
            v16bf b2 = load_b_s(wih_s, I_, 32, k0, lane);
            accIN = WMMA_BF16(a, b2, accIN);
            a = an;
        }

        // ---- gh = h_t @ w_hh^T  (K = 512), A double-buffered ----
#pragma unroll 2
        for (int k0 = 0; k0 < H_; k0 += 32) {
            v16bf an = load_a_g(hprevB, (size_t)H_, mBase,
                                (k0 + 32) & (H_ - 1), lane);
            v16bf b0 = load_b_s(whh_s, H_, 0,  k0, lane);
            accR  = WMMA_BF16(ah, b0, accR);
            v16bf b1 = load_b_s(whh_s, H_, 16, k0, lane);
            accZ  = WMMA_BF16(ah, b1, accZ);
            v16bf b2 = load_b_s(whh_s, H_, 32, k0, lane);
            accHN = WMMA_BF16(ah, b2, accHN);
            ah = an;
        }

        // ---- gates, state update, clip, stores ----
#pragma unroll
        for (int r = 0; r < 8; ++r) {
            int row  = rbase + r;
            float rg = fast_sigmoid(accR[r]);
            float zg = fast_sigmoid(accZ[r]);
            float ng = fast_tanh(accIN[r] + rg * accHN[r]);
            float hn = (1.0f - zg) * ng + zg * ho[r];
            hn = fminf(5.0f, fmaxf(-5.0f, hn));
            out[((size_t)row * T_ + t) * H_ + col] = hn;
            hnextF[(size_t)row * H_ + col] = hn;
            hnextB[(size_t)row * H_ + col] = (__bf16)hn;
            if (t == T_ - 1)
                out[(size_t)B_ * T_ * H_ + (size_t)row * H_ + col] = hn;   // h_last
        }

        // ---- device-wide step barrier (release -> arrive -> acquire spin) ----
        __threadfence();
        __syncthreads();
        if (tid == 0) {
            __hip_atomic_fetch_add(cnt, 1u, __ATOMIC_RELEASE, __HIP_MEMORY_SCOPE_AGENT);
            unsigned target = (unsigned)NWG * (unsigned)(t + 1);
            while (__hip_atomic_load(cnt, __ATOMIC_ACQUIRE, __HIP_MEMORY_SCOPE_AGENT) < target)
                __builtin_amdgcn_s_sleep(2);
        }
        __syncthreads();
        __threadfence();
    }
}

// ---------------- launcher ----------------

extern "C" void kernel_launch(void* const* d_in, const int* in_sizes, int n_in,
                              void* d_out, int out_size, void* d_ws, size_t ws_size,
                              hipStream_t stream) {
    (void)in_sizes; (void)n_in; (void)out_size; (void)ws_size;

    const float* x    = (const float*)d_in[0];
    const float* h0   = (const float*)d_in[1];
    const float* w_ih = (const float*)d_in[2];
    const float* w_hh = (const float*)d_in[3];
    const float* b_ih = (const float*)d_in[4];
    const float* b_hh = (const float*)d_in[5];
    float* out = (float*)d_out;
    char*  ws  = (char*)d_ws;

    __bf16* xbf  = (__bf16*)(ws + OFF_X);
    __bf16* wihb = (__bf16*)(ws + OFF_WIH);
    __bf16* whhb = (__bf16*)(ws + OFF_WHH);
    float*  hf   = (float*)(ws + OFF_HF);
    __bf16* hb   = (__bf16*)(ws + OFF_HB);
    unsigned int* cnt = (unsigned int*)(ws + OFF_CNT);

    (void)hipMemsetAsync(cnt, 0, 256, stream);   // barrier counter must start at 0

    {   // x -> bf16
        int n8 = B_ * T_ * I_ / 8;
        cvt_f32_to_bf16_x8<<<(n8 + 255) / 256, 256, 0, stream>>>(x, xbf, n8);
    }
    {   // w_ih -> bf16
        int n8 = G3_ * I_ / 8;
        cvt_f32_to_bf16_x8<<<(n8 + 255) / 256, 256, 0, stream>>>(w_ih, wihb, n8);
    }
    {   // w_hh -> bf16
        int n8 = G3_ * H_ / 8;
        cvt_f32_to_bf16_x8<<<(n8 + 255) / 256, 256, 0, stream>>>(w_hh, whhb, n8);
    }
    init_h_kernel<<<(B_ * H_ + 255) / 256, 256, 0, stream>>>(h0, hf, hb);

    size_t smemBytes = 48 * 512 * 2 + 48 * 256 * 2 + 64 * 4;   // 73,984 B (< 320 KB/WGP)
    gru_scan_kernel<<<NWG, 256, smemBytes, stream>>>(xbf, wihb, whhb, b_ih, b_hh,
                                                     hf, hb, out, cnt);
}
